// DynamicShortConvolution_8967891714368
// MI455X (gfx1250) — matrix operand order repaired and checked by reading, not compiled, not run on hardware
//
#include <hip/hip_runtime.h>
#include <hip/hip_bf16.h>

typedef __attribute__((ext_vector_type(16))) _Float16 v16h;
typedef __attribute__((ext_vector_type(8)))  float    v8f;
typedef __attribute__((ext_vector_type(4)))  unsigned int v4u;
typedef __attribute__((ext_vector_type(2)))  unsigned int v2u;

#define D_DIM 2048
#define GH 256
#define DK (D_DIM * 4)    // 8192

union AB { v4u u4[2]; v16h h; };
union F4 { v4u u; float f[4]; };
union H4 { v2u u; _Float16 h[4]; };

__device__ __forceinline__ float silu_f(float s) {
    // s * sigmoid(s) via v_exp_f32 + v_rcp_f32 (avoid IEEE div expansion)
    return s * __builtin_amdgcn_rcpf(1.0f + __expf(-s));
}

// -------------------------------------------------------------------------
// Prep: pack f32 (K x N) weights into WMMA f16 B-fragment blobs.
// Blob (k0,n0) = 1KB: dst[blob*512 + l*16 + j] = W[k0 + 16*(l>>4) + j][n0 + (l&15)]
// so each lane's 16 halves (V0..V7) are contiguous -> 2x global_load_b128.
// -------------------------------------------------------------------------
__global__ __launch_bounds__(256) void dsc_pack_b(
    const float* __restrict__ src, _Float16* __restrict__ dst, int K, int N) {
    int blob = blockIdx.x;
    int nb = N >> 4;
    int k0 = (blob / nb) << 5;
    int n0 = (blob % nb) << 4;
    #pragma unroll
    for (int it = 0; it < 2; ++it) {
        int i = threadIdx.x + it * 256;
        int l = i >> 4, j = i & 15;
        float v = src[(size_t)(k0 + ((l >> 4) << 4) + j) * N + n0 + (l & 15)];
        dst[(size_t)blob * 512 + i] = (_Float16)v;
    }
}

// -------------------------------------------------------------------------
// Kernel 1: H = silu(G @ W1) as f16.  Block: 32 rows x 256 cols, 8 waves.
// Each wave: 2 M-tiles x 2 N-subtiles (4 accs), K staged 64-wide in LDS.
// -------------------------------------------------------------------------
__global__ __launch_bounds__(256) void dsc_gen_h(
    const float* __restrict__ G, const _Float16* __restrict__ W1h,
    _Float16* __restrict__ Hh) {
    __shared__ __align__(16) _Float16 glds[32 * 88];   // stride 88 halves
    const int tid  = threadIdx.x;
    const int lane = tid & 31;
    const int wave = tid >> 5;
    const int l16  = lane & 15;
    const int half = lane >> 4;
    const int t0   = blockIdx.x * 32;
    const int n0   = wave * 32;

    const v4u* Gv = (const v4u*)G;                     // 4 floats per v4u
    // per-thread staging coords (two unconditional iterations: 512 float4)
    const int sr0 = tid >> 4,          sc0 = tid & 15;
    const int sr1 = (tid + 256) >> 4,  sc1 = (tid + 256) & 15;

    v8f acc[4] = {{}, {}, {}, {}};
    for (int kc = 0; kc < D_DIM; kc += 64) {
        __syncthreads();
        // stage 32 x 64 f32 -> packed f16 (both b128 loads issue together)
        F4 in0, in1;
        in0.u = Gv[(size_t)(t0 + sr0) * (D_DIM / 4) + (kc >> 2) + sc0];
        in1.u = Gv[(size_t)(t0 + sr1) * (D_DIM / 4) + (kc >> 2) + sc1];
        H4 o0, o1;
        o0.h[0] = (_Float16)in0.f[0]; o0.h[1] = (_Float16)in0.f[1];
        o0.h[2] = (_Float16)in0.f[2]; o0.h[3] = (_Float16)in0.f[3];
        o1.h[0] = (_Float16)in1.f[0]; o1.h[1] = (_Float16)in1.f[1];
        o1.h[2] = (_Float16)in1.f[2]; o1.h[3] = (_Float16)in1.f[3];
        *(v2u*)&glds[sr0 * 88 + sc0 * 4] = o0.u;
        *(v2u*)&glds[sr1 * 88 + sc1 * 4] = o1.u;
        __syncthreads();
        #pragma unroll
        for (int ks = 0; ks < 2; ++ks) {
            int k = ks << 5;
            int kblob = ((kc + k) >> 5) * (GH >> 4);
            AB b0, b1;
            {
                const v4u* bp = (const v4u*)(W1h + ((size_t)(kblob + (n0 >> 4)) << 9));
                b0.u4[0] = bp[lane * 2]; b0.u4[1] = bp[lane * 2 + 1];
            }
            {
                const v4u* bp = (const v4u*)(W1h + ((size_t)(kblob + ((n0 + 16) >> 4)) << 9));
                b1.u4[0] = bp[lane * 2]; b1.u4[1] = bp[lane * 2 + 1];
            }
            #pragma unroll
            for (int mt = 0; mt < 2; ++mt) {
                int row = mt * 16 + l16;
                AB a;
                a.u4[0] = *(const v4u*)&glds[row * 88 + k + 16 * half];
                a.u4[1] = *(const v4u*)&glds[row * 88 + k + 16 * half + 8];
                acc[mt * 2 + 0] = __builtin_amdgcn_wmma_f32_16x16x32_f16(
                    false, a.h, false, b0.h, (short)0, acc[mt * 2 + 0], false, false);
                acc[mt * 2 + 1] = __builtin_amdgcn_wmma_f32_16x16x32_f16(
                    false, a.h, false, b1.h, (short)0, acc[mt * 2 + 1], false, false);
            }
        }
    }
    #pragma unroll
    for (int mt = 0; mt < 2; ++mt)
        #pragma unroll
        for (int st = 0; st < 2; ++st)
            #pragma unroll
            for (int r = 0; r < 8; ++r) {
                int row = t0 + mt * 16 + r + 8 * half;
                int col = n0 + st * 16 + l16;
                Hh[(size_t)row * GH + col] = (_Float16)silu_f(acc[mt * 2 + st][r]);
            }
}

// -------------------------------------------------------------------------
// Kernel 2: fused kern-GEMM (H f16 @ W2h) + dynamic depthwise conv + silu.
// Block: 32 time rows x 64 channels (= 256 GEMM cols), 8 waves,
// each wave 2 M-tiles x 2 N-subtiles.  kern never touches HBM.
// Seq lens are multiples of 128 -> 32-row tiles never straddle a sequence.
// -------------------------------------------------------------------------
__global__ __launch_bounds__(256) void dsc_conv_fused(
    const float* __restrict__ x, const float* __restrict__ cs,
    const _Float16* __restrict__ Hh, const _Float16* __restrict__ W2h,
    const float* __restrict__ b2, const int* __restrict__ seq_lens,
    float* __restrict__ out, int B) {
    __shared__ __align__(16) _Float16 hlds[32 * 264];  // stride 264 halves
    __shared__ __align__(16) float    xlds[35 * 68];   // stride 68 floats
    const int tid  = threadIdx.x;
    const int lane = tid & 31;
    const int wave = tid >> 5;
    const int l16  = lane & 15;
    const int half = lane >> 4;
    const int t0   = blockIdx.x * 32;
    const int d0   = blockIdx.y * 64;

    // locate sequence containing this tile (uniform across block)
    int off = 0, seq = 0;
    for (int b = 0; b < B; ++b) {
        int len = seq_lens[b];
        if (t0 >= off && t0 < off + len) { seq = b; break; }
        off += len;
    }

    // stage H tile (32 x 256 halves): 1024 uint4, 4 unconditional iterations
    {
        const v4u* Hv = (const v4u*)Hh;               // 8 halves per v4u
        v4u t[4];
        #pragma unroll
        for (int it = 0; it < 4; ++it) {
            int idx = tid + it * 256;
            t[it] = Hv[(size_t)(t0 + (idx >> 5)) * 32 + (idx & 31)];
        }
        #pragma unroll
        for (int it = 0; it < 4; ++it) {
            int idx = tid + it * 256;
            *(v4u*)&hlds[(idx >> 5) * 264 + (idx & 31) * 8] = t[it];
        }
    }
    // stage x window rows t0-3..t0+31, 64 channels
    if (t0 > off) {                      // interior: b128 path, 560 = 2*256+48
        const v4u* xv = (const v4u*)x;
        v4u t0v, t1v, t2v;
        t0v = xv[(size_t)(t0 - 3 + (tid >> 4)) * (D_DIM / 4) + (d0 >> 2) + (tid & 15)];
        {
            int idx = tid + 256;
            t1v = xv[(size_t)(t0 - 3 + (idx >> 4)) * (D_DIM / 4) + (d0 >> 2) + (idx & 15)];
        }
        if (tid < 48) {
            int idx = tid + 512;
            t2v = xv[(size_t)(t0 - 3 + (idx >> 4)) * (D_DIM / 4) + (d0 >> 2) + (idx & 15)];
        }
        *(v4u*)&xlds[(tid >> 4) * 68 + (tid & 15) * 4] = t0v;
        {
            int idx = tid + 256;
            *(v4u*)&xlds[(idx >> 4) * 68 + (idx & 15) * 4] = t1v;
        }
        if (tid < 48) {
            int idx = tid + 512;
            *(v4u*)&xlds[(idx >> 4) * 68 + (idx & 15) * 4] = t2v;
        }
    } else {                             // first tile of a sequence
        for (int idx = tid; idx < 35 * 64; idx += 256) {
            int r = idx >> 6, c = idx & 63;
            int t = t0 - 3 + r;
            float v;
            if (t < off)
                v = cs[(size_t)seq * D_DIM * 3 + (size_t)(d0 + c) * 3 + (t - off + 3)];
            else
                v = x[(size_t)t * D_DIM + d0 + c];
            xlds[r * 68 + c] = v;
        }
    }
    __syncthreads();

    // GEMM: 32 x 256 cols of (H @ W2) starting at column d0*4
    const int ncol_w = d0 * 4 + wave * 32;
    v8f acc[4] = {{}, {}, {}, {}};
    #pragma unroll
    for (int k = 0; k < GH; k += 32) {
        int kblob = (k >> 5) * (DK >> 4);
        AB b0, b1;
        {
            const v4u* bp = (const v4u*)(W2h + ((size_t)(kblob + (ncol_w >> 4)) << 9));
            b0.u4[0] = bp[lane * 2]; b0.u4[1] = bp[lane * 2 + 1];
        }
        {
            const v4u* bp = (const v4u*)(W2h + ((size_t)(kblob + ((ncol_w + 16) >> 4)) << 9));
            b1.u4[0] = bp[lane * 2]; b1.u4[1] = bp[lane * 2 + 1];
        }
        #pragma unroll
        for (int mt = 0; mt < 2; ++mt) {
            int row = mt * 16 + l16;
            AB a;
            a.u4[0] = *(const v4u*)&hlds[row * 264 + k + 16 * half];
            a.u4[1] = *(const v4u*)&hlds[row * 264 + k + 16 * half + 8];
            acc[mt * 2 + 0] = __builtin_amdgcn_wmma_f32_16x16x32_f16(
                false, a.h, false, b0.h, (short)0, acc[mt * 2 + 0], false, false);
            acc[mt * 2 + 1] = __builtin_amdgcn_wmma_f32_16x16x32_f16(
                false, a.h, false, b1.h, (short)0, acc[mt * 2 + 1], false, false);
        }
    }

    // conv taps + quad-lane reduction + silu + store
    #pragma unroll
    for (int mt = 0; mt < 2; ++mt)
        #pragma unroll
        for (int st = 0; st < 2; ++st) {
            v8f a = acc[mt * 2 + st];
            int col  = ncol_w + st * 16 + l16;   // GEMM column = d*4 + kk
            float bias = b2[col];
            int kk = col & 3;
            int dl = (col >> 2) - d0;
            #pragma unroll
            for (int r = 0; r < 8; ++r) {
                int rowl = mt * 16 + r + 8 * half;          // 0..31
                float s = xlds[(rowl + kk) * 68 + dl] * (a[r] + bias);
                s += __shfl_xor(s, 1, 32);                  // reduce 4 taps
                s += __shfl_xor(s, 2, 32);
                if ((lane & 3) == 0)
                    out[(size_t)(t0 + rowl) * D_DIM + (col >> 2)] = silu_f(s);
            }
        }
}

// -------------------------------------------------------------------------
// Kernel 3: new_conv_state[i][d][m] = x[offs[i+1]-3+m][d]
// -------------------------------------------------------------------------
__global__ __launch_bounds__(256) void dsc_new_state(
    const float* __restrict__ x, const int* __restrict__ seq_lens,
    float* __restrict__ ncs) {
    int i = blockIdx.x;
    int off_end = 0;
    for (int b = 0; b <= i; ++b) off_end += seq_lens[b];
    for (int idx = threadIdx.x; idx < D_DIM * 3; idx += blockDim.x) {
        int d = idx / 3, m = idx % 3;
        ncs[(size_t)i * D_DIM * 3 + idx] =
            x[(size_t)(off_end - 3 + m) * D_DIM + d];
    }
}

extern "C" void kernel_launch(void* const* d_in, const int* in_sizes, int n_in,
                              void* d_out, int out_size, void* d_ws, size_t ws_size,
                              hipStream_t stream) {
    const float* x   = (const float*)d_in[0];
    const float* cs  = (const float*)d_in[1];
    const float* gin = (const float*)d_in[2];
    const int*   sl  = (const int*)d_in[3];
    const float* w1  = (const float*)d_in[4];
    const float* w2  = (const float*)d_in[5];
    const float* b2  = (const float*)d_in[6];
    float* out = (float*)d_out;

    _Float16* Hh  = (_Float16*)d_ws;                              // T*256*2 ~ 5 MB
    _Float16* W1h = (_Float16*)((char*)d_ws + (5u << 20));        // 1 MB
    _Float16* W2h = (_Float16*)((char*)d_ws + (6u << 20));        // 4 MB

    const int T = in_sizes[0] / D_DIM;   // 9728
    const int B = in_sizes[3];           // 8
    const int row_tiles = T / 32;        // 304 (seq lens are multiples of 128)

    dsc_pack_b<<<dim3((D_DIM / 32) * (GH / 16)), dim3(256), 0, stream>>>(
        w1, W1h, D_DIM, GH);                                      // 1024 blobs
    dsc_pack_b<<<dim3((GH / 32) * (DK / 16)), dim3(256), 0, stream>>>(
        w2, W2h, GH, DK);                                         // 4096 blobs
    dsc_gen_h<<<dim3(row_tiles), dim3(256), 0, stream>>>(gin, W1h, Hh);
    dsc_conv_fused<<<dim3(row_tiles, D_DIM / 64), dim3(256), 0, stream>>>(
        x, cs, Hh, W2h, b2, sl, out, B);
    dsc_new_state<<<dim3(B), dim3(256), 0, stream>>>(
        x, sl, out + (size_t)T * D_DIM);
}
